// MHALayer_6940667151026
// MI455X (gfx1250) — compile-verified, matrix operand order checked by reading, not run on hardware
//
#include <hip/hip_runtime.h>

// ---------------------------------------------------------------------------
// MHA layer for gfx1250 (MI455X): bf16 WMMA GEMMs with async-to-LDS weight
// staging (GLOBAL_LOAD_ASYNC_TO_LDS_B128 / ASYNCcnt double buffering).
// B=4, S=2048, HID=512, HEADS=8, HD=64
// ---------------------------------------------------------------------------

typedef __attribute__((ext_vector_type(16))) __bf16 v16bf;
typedef __attribute__((ext_vector_type(8)))  float  v8f;

#define WMMA_BF16(a, b, c) \
    __builtin_amdgcn_wmma_f32_16x16x32_bf16(false, (a), false, (b), (short)0, (c), false, false)

#define S_LEN   2048
#define HIDD    512
#define NHEAD   8
#define HDIM    64
#define NEGINF  (-1e10f)
#define KSTEPS  (HIDD / 32)                              // 16 K-steps

// Load 16 bf16 elems in WMMA A/B K-striping (two 16-byte loads):
//   elems 0..7  <- p[koff .. koff+7]
//   elems 8..15 <- p[16+koff .. 23+koff]
__device__ inline v16bf ld16_bf(const __bf16* p, int koff) {
    v16bf v;
#pragma unroll
    for (int i = 0; i < 8; ++i) v[i] = p[koff + i];
#pragma unroll
    for (int i = 0; i < 8; ++i) v[8 + i] = p[16 + koff + i];
    return v;
}

// Async-stage one 64col x 32k bf16 W tile (4 KB) into LDS.
// 256 16-byte transfers; 128 threads x 2 -> each wave issues exactly 2
// global_load_async_to_lds_b128 (ASYNCcnt += 2 per wave).
__device__ inline void stage_w(const __bf16* __restrict__ W, int col0, int kb,
                               __bf16* buf, int tid) {
#pragma unroll
    for (int i = 0; i < 2; ++i) {
        const int t   = tid + i * 128;                   // 0..255
        const int row = t >> 2;                          // 0..63 (column of W^T)
        const int q   = t & 3;                           // 16B quarter of 64B row
        const __bf16* g = W + (size_t)(col0 + row) * HIDD + kb + q * 8;
        const unsigned lds = (unsigned)(size_t)(buf + row * 32 + q * 8);
        asm volatile("global_load_async_to_lds_b128 %0, %1, off"
                     :: "v"(lds), "v"(g) : "memory");
    }
}

__device__ inline void wait_async_le2() {
    asm volatile("s_wait_asynccnt 0x2" ::: "memory");
}
__device__ inline void wait_async_0() {
    asm volatile("s_wait_asynccnt 0x0" ::: "memory");
}

// ---------------------------------------------------------------------------
// Kernel 0: fp32 -> bf16 conversion (one-time).
// ---------------------------------------------------------------------------
__global__ __launch_bounds__(256) void cvt_bf16_kernel(
    const float* __restrict__ in, __bf16* __restrict__ out, int n) {
    const int i = (blockIdx.x * 256 + threadIdx.x) * 8;
    if (i + 8 <= n) {
#pragma unroll
        for (int j = 0; j < 8; ++j) out[i + j] = (__bf16)in[i + j];
    }
}

// ---------------------------------------------------------------------------
// Kernel 1: fused Q/K/V projection.  Y = X @ W^T + b (bf16 operands).
// Block = 4 waves sharing one (which, nt) -> W tiles staged once in LDS via
// async copies, double-buffered.  Wave = 32(M) x 64(N) tile.
// Grid: 3 which * 8 nt * 64 mt-groups = 1536 blocks.
// Q,K stored bf16 [B,H,S,HD]; V stored TRANSPOSED bf16 [B,H,HD,S].
// ---------------------------------------------------------------------------
__global__ __launch_bounds__(128) void qkv_proj_kernel(
    const __bf16* __restrict__ X0, const __bf16* __restrict__ X1, const __bf16* __restrict__ X2,
    const __bf16* __restrict__ W0, const __bf16* __restrict__ W1, const __bf16* __restrict__ W2,
    const float*  __restrict__ B0, const float*  __restrict__ B1, const float*  __restrict__ B2,
    __bf16* __restrict__ Oq, __bf16* __restrict__ Ok, __bf16* __restrict__ Ovt) {
    __shared__ __bf16 wlds[2][64 * 32];                  // 2 x 4 KB double buffer

    const int tid    = threadIdx.x;
    const int lane   = tid & 31;
    const int wave   = tid >> 5;
    const int which  = blockIdx.x >> 9;                  // 0..2
    const int rem    = blockIdx.x & 511;
    const int nt     = rem >> 6;                         // 0..7 -> 64-col group
    const int mt     = (rem & 63) * 4 + wave;            // 0..255 -> 32-row group
    const int lane15 = lane & 15;
    const int half   = lane >> 4;
    const int koff   = half * 8;
    const int row0   = mt * 32;
    const int col0   = nt * 64;

    const __bf16* X  = (which == 0) ? X0 : (which == 1) ? X1 : X2;
    const __bf16* W  = (which == 0) ? W0 : (which == 1) ? W1 : W2;
    const float*  Bv = (which == 0) ? B0 : (which == 1) ? B1 : B2;

    const __bf16* x0 = X + (row0 + lane15) * HIDD;       // A tile 0: rows 0..15
    const __bf16* x1 = X + (row0 + 16 + lane15) * HIDD;  // A tile 1: rows 16..31

    v8f acc[2][4] = {};

    stage_w(W, col0, 0, wlds[0], tid);                   // prologue stage
    for (int i = 0; i < KSTEPS - 1; ++i) {
        stage_w(W, col0, (i + 1) * 32, wlds[(i + 1) & 1], tid);
        wait_async_le2();                                // stage i complete
        __syncthreads();
        const int kb = i * 32;
        const __bf16* wb = wlds[i & 1];
        v16bf a0 = ld16_bf(x0 + kb, koff);
        v16bf a1 = ld16_bf(x1 + kb, koff);
#pragma unroll
        for (int n = 0; n < 4; ++n) {
            v16bf bm = ld16_bf(wb + (n * 16 + lane15) * 32, koff);
            acc[0][n] = WMMA_BF16(a0, bm, acc[0][n]);
            acc[1][n] = WMMA_BF16(a1, bm, acc[1][n]);
        }
        __syncthreads();                                 // done reading buf i&1
    }
    {
        wait_async_0();
        __syncthreads();
        const int kb = (KSTEPS - 1) * 32;
        const __bf16* wb = wlds[(KSTEPS - 1) & 1];
        v16bf a0 = ld16_bf(x0 + kb, koff);
        v16bf a1 = ld16_bf(x1 + kb, koff);
#pragma unroll
        for (int n = 0; n < 4; ++n) {
            v16bf bm = ld16_bf(wb + (n * 16 + lane15) * 32, koff);
            acc[0][n] = WMMA_BF16(a0, bm, acc[0][n]);
            acc[1][n] = WMMA_BF16(a1, bm, acc[1][n]);
        }
    }

#pragma unroll
    for (int n = 0; n < 4; ++n) {
        const int   col  = col0 + n * 16 + lane15;
        const float bias = Bv[col];
        const int   h    = col >> 6;
        const int   d    = col & 63;
#pragma unroll
        for (int m = 0; m < 2; ++m) {
#pragma unroll
            for (int r = 0; r < 8; ++r) {
                const int row = row0 + m * 16 + r + 8 * half;  // C: M = r + 8*half
                const int bb  = row >> 11;
                const int s   = row & 2047;
                const float val = acc[m][n][r] + bias;
                if (which == 2) {
                    // V transposed: [B,H,HD,S]  (lane-contiguous along s)
                    Ovt[((size_t)(bb * NHEAD + h) * HDIM + d) * S_LEN + s] = (__bf16)val;
                } else {
                    __bf16* O = (which == 0) ? Oq : Ok;
                    O[((size_t)(bb * NHEAD + h) * S_LEN + s) * HDIM + d] = (__bf16)val;
                }
            }
        }
    }
}

// ---------------------------------------------------------------------------
// Kernel 2: flash-style attention.  One wave = 16 query rows of one (b,h).
// 32 * 128 = 4096 waves, 4 waves/block, per-wave 16x32 bf16 LDS P-tile.
// ---------------------------------------------------------------------------
__global__ __launch_bounds__(128) void attn_kernel(
    const __bf16* __restrict__ Qws, const __bf16* __restrict__ Kws,
    const __bf16* __restrict__ Vtws, const int* __restrict__ mask,
    __bf16* __restrict__ attn) {
    __shared__ __bf16 plds[4][16 * 32];

    const int lane   = threadIdx.x & 31;
    const int wave   = threadIdx.x >> 5;
    const int gid    = blockIdx.x * 4 + wave;            // 0 .. 4095
    const int bh     = gid >> 7;                         // b*8 + h
    const int qb     = gid & 127;                        // 16-row query block
    const int b_     = bh >> 3;
    const int lane15 = lane & 15;
    const int half   = lane >> 4;
    const int koff   = half * 8;

    const __bf16* Qp  = Qws  + (size_t)bh * S_LEN * HDIM;
    const __bf16* Kp  = Kws  + (size_t)bh * S_LEN * HDIM;
    const __bf16* Vtp = Vtws + (size_t)bh * HDIM * S_LEN;
    const int*    mp  = mask + b_ * S_LEN;
    __bf16*       pl  = plds[wave];

    const __bf16* qrow = Qp + (qb * 16 + lane15) * HDIM;
    const v16bf aq0 = ld16_bf(qrow, koff);
    const v16bf aq1 = ld16_bf(qrow + 32, koff);

    float rmax[8], rsum[8];
#pragma unroll
    for (int r = 0; r < 8; ++r) { rmax[r] = -INFINITY; rsum[r] = 0.0f; }
    v8f o[4] = {};

    const float scale = 0.125f;                          // 1/sqrt(64)

    for (int kb = 0; kb < S_LEN; kb += 32) {
        // --- E = Q K^T for 32 keys (two 16-wide N tiles) -------------------
        const __bf16* krow0 = Kp + (kb + lane15) * HDIM;
        const __bf16* krow1 = Kp + (kb + 16 + lane15) * HDIM;
        v8f e0 = {}, e1 = {};
        e0 = WMMA_BF16(aq0, ld16_bf(krow0,      koff), e0);
        e0 = WMMA_BF16(aq1, ld16_bf(krow0 + 32, koff), e0);
        e1 = WMMA_BF16(aq0, ld16_bf(krow1,      koff), e1);
        e1 = WMMA_BF16(aq1, ld16_bf(krow1 + 32, koff), e1);

        const int mk0 = mp[kb + lane15];
        const int mk1 = mp[kb + 16 + lane15];

        // --- online softmax per row (row state replicated across 16 lanes) -
        float alpha[8];
#pragma unroll
        for (int r = 0; r < 8; ++r) {
            float x0 = (mk0 == 0) ? NEGINF : e0[r] * scale;
            float x1 = (mk1 == 0) ? NEGINF : e1[r] * scale;
            float bm = fmaxf(x0, x1);
#pragma unroll
            for (int off = 1; off < 16; off <<= 1)
                bm = fmaxf(bm, __shfl_xor(bm, off, 32));
            const float nm = fmaxf(rmax[r], bm);
            alpha[r] = __expf(rmax[r] - nm);
            rmax[r]  = nm;
            const float p0 = __expf(x0 - nm);
            const float p1 = __expf(x1 - nm);
            float ps = p0 + p1;
#pragma unroll
            for (int off = 1; off < 16; off <<= 1)
                ps += __shfl_xor(ps, off, 32);
            rsum[r] = rsum[r] * alpha[r] + ps;
            const int prow = r + 8 * half;               // C-layout row
            pl[prow * 32 + lane15]      = (__bf16)p0;
            pl[prow * 32 + 16 + lane15] = (__bf16)p1;
        }

        // --- C-layout -> A-layout via LDS (same-wave DS ops are in-order) --
        const __bf16* pr = pl + lane15 * 32;
        const v16bf ap = ld16_bf(pr, koff);

        // --- O += P x V, alpha rescale folded into C -----------------------
#pragma unroll
        for (int dt = 0; dt < 4; ++dt) {
            const __bf16* vcol = Vtp + (size_t)(dt * 16 + lane15) * S_LEN + kb;
            v16bf bv = ld16_bf(vcol, koff);
            v8f c;
#pragma unroll
            for (int r = 0; r < 8; ++r) c[r] = o[dt][r] * alpha[r];
            o[dt] = WMMA_BF16(ap, bv, c);
        }
    }

    // --- epilogue: normalize and store to [B,S,HID] bf16 -------------------
    const int hcol = (bh & 7) * HDIM;
#pragma unroll
    for (int dt = 0; dt < 4; ++dt) {
        const int col = hcol + dt * 16 + lane15;
#pragma unroll
        for (int r = 0; r < 8; ++r) {
            const int s = qb * 16 + r + 8 * half;
            attn[((size_t)b_ * S_LEN + s) * HIDD + col] = (__bf16)(o[dt][r] / rsum[r]);
        }
    }
}

// ---------------------------------------------------------------------------
// Kernel 3: output projection.  out = A @ Wo^T + bo (bf16 ops, fp32 result).
// Same async W-staging structure.  Grid: 8 nt * 64 mt-groups = 512 blocks.
// ---------------------------------------------------------------------------
__global__ __launch_bounds__(128) void out_proj_kernel(
    const __bf16* __restrict__ X, const __bf16* __restrict__ W,
    const float* __restrict__ Bv, float* __restrict__ out) {
    __shared__ __bf16 wlds[2][64 * 32];

    const int tid    = threadIdx.x;
    const int lane   = tid & 31;
    const int wave   = tid >> 5;
    const int nt     = blockIdx.x >> 6;                  // 0..7
    const int mt     = (blockIdx.x & 63) * 4 + wave;     // 0..255
    const int lane15 = lane & 15;
    const int half   = lane >> 4;
    const int koff   = half * 8;
    const int row0   = mt * 32;
    const int col0   = nt * 64;

    const __bf16* x0 = X + (row0 + lane15) * HIDD;
    const __bf16* x1 = X + (row0 + 16 + lane15) * HIDD;

    v8f acc[2][4] = {};

    stage_w(W, col0, 0, wlds[0], tid);
    for (int i = 0; i < KSTEPS - 1; ++i) {
        stage_w(W, col0, (i + 1) * 32, wlds[(i + 1) & 1], tid);
        wait_async_le2();
        __syncthreads();
        const int kb = i * 32;
        const __bf16* wb = wlds[i & 1];
        v16bf a0 = ld16_bf(x0 + kb, koff);
        v16bf a1 = ld16_bf(x1 + kb, koff);
#pragma unroll
        for (int n = 0; n < 4; ++n) {
            v16bf bm = ld16_bf(wb + (n * 16 + lane15) * 32, koff);
            acc[0][n] = WMMA_BF16(a0, bm, acc[0][n]);
            acc[1][n] = WMMA_BF16(a1, bm, acc[1][n]);
        }
        __syncthreads();
    }
    {
        wait_async_0();
        __syncthreads();
        const int kb = (KSTEPS - 1) * 32;
        const __bf16* wb = wlds[(KSTEPS - 1) & 1];
        v16bf a0 = ld16_bf(x0 + kb, koff);
        v16bf a1 = ld16_bf(x1 + kb, koff);
#pragma unroll
        for (int n = 0; n < 4; ++n) {
            v16bf bm = ld16_bf(wb + (n * 16 + lane15) * 32, koff);
            acc[0][n] = WMMA_BF16(a0, bm, acc[0][n]);
            acc[1][n] = WMMA_BF16(a1, bm, acc[1][n]);
        }
    }

#pragma unroll
    for (int n = 0; n < 4; ++n) {
        const int   col  = col0 + n * 16 + lane15;
        const float bias = Bv[col];
#pragma unroll
        for (int m = 0; m < 2; ++m) {
#pragma unroll
            for (int r = 0; r < 8; ++r) {
                const int row = row0 + m * 16 + r + 8 * half;
                out[(size_t)row * HIDD + col] = acc[m][n][r] + bias;
            }
        }
    }
}

// ---------------------------------------------------------------------------
extern "C" void kernel_launch(void* const* d_in, const int* in_sizes, int n_in,
                              void* d_out, int out_size, void* d_ws, size_t ws_size,
                              hipStream_t stream) {
    const float* q    = (const float*)d_in[0];
    const float* k    = (const float*)d_in[1];
    const float* v    = (const float*)d_in[2];
    const int*   mask = (const int*)  d_in[3];
    const float* Wq   = (const float*)d_in[4];
    const float* Wk   = (const float*)d_in[5];
    const float* Wv   = (const float*)d_in[6];
    const float* Wo   = (const float*)d_in[7];
    const float* bq   = (const float*)d_in[8];
    const float* bk   = (const float*)d_in[9];
    const float* bv   = (const float*)d_in[10];
    const float* bo   = (const float*)d_in[11];

    const size_t NX = (size_t)4 * S_LEN * HIDD;          // 4,194,304 elems
    const size_t NW = (size_t)HIDD * HIDD;               //   262,144 elems

    __bf16* ws   = (__bf16*)d_ws;
    __bf16* Xq   = ws;                                   // bf16 inputs
    __bf16* Xk   = Xq + NX;
    __bf16* Xv   = Xk + NX;
    __bf16* Wqb  = Xv + NX;                              // bf16 weights
    __bf16* Wkb  = Wqb + NW;
    __bf16* Wvb  = Wkb + NW;
    __bf16* Wob  = Wvb + NW;
    __bf16* Qws  = Wob + NW;                             // Q [B,H,S,HD]
    __bf16* Kws  = Qws + NX;                             // K [B,H,S,HD]
    __bf16* Vtws = Kws + NX;                             // V [B,H,HD,S]
    __bf16* Aws  = Vtws + NX;                            // attn out [B,S,HID]

    const int xb = (int)(NX / 8 / 256);                  // 2048 blocks
    const int wb = (int)(NW / 8 / 256);                  //  128 blocks
    cvt_bf16_kernel<<<xb, 256, 0, stream>>>(q,  Xq,  (int)NX);
    cvt_bf16_kernel<<<xb, 256, 0, stream>>>(k,  Xk,  (int)NX);
    cvt_bf16_kernel<<<xb, 256, 0, stream>>>(v,  Xv,  (int)NX);
    cvt_bf16_kernel<<<wb, 256, 0, stream>>>(Wq, Wqb, (int)NW);
    cvt_bf16_kernel<<<wb, 256, 0, stream>>>(Wk, Wkb, (int)NW);
    cvt_bf16_kernel<<<wb, 256, 0, stream>>>(Wv, Wvb, (int)NW);
    cvt_bf16_kernel<<<wb, 256, 0, stream>>>(Wo, Wob, (int)NW);

    qkv_proj_kernel<<<1536, 128, 0, stream>>>(Xq, Xk, Xv, Wqb, Wkb, Wvb,
                                              bq, bk, bv, Qws, Kws, Vtws);
    attn_kernel<<<1024, 128, 0, stream>>>(Qws, Kws, Vtws, mask, Aws);
    out_proj_kernel<<<512, 128, 0, stream>>>(Aws, Wob, bo, (float*)d_out);
}
